// CapsNet_68624987456208
// MI455X (gfx1250) — matrix operand order repaired and checked by reading, not compiled, not run on hardware
//
#include <hip/hip_runtime.h>
#include <math.h>

// MI455X (gfx1250) CapsNet forward. wave32, WMMA f16 -> f32 accumulate.
// Heavy GEMMs (primary-caps implicit conv, decoder layers) use
// v_wmma_f32_16x16x32_f16 with LDS-staged f16 tiles; fp32 weights are
// converted in-flight (dec_w3 is 420MB fp32 -> reading it once at 23.3TB/s
// is the floor; a separate conversion pass would only add traffic).
//
// Grid mapping note: in the dense GEMM, blockIdx.x = m-tile (only 8 of them)
// so concurrently-scheduled blocks share the same weight tile and the 420MB
// dec_w3 stream is fetched from HBM once (7/8 reads hit L2) instead of 8x.

typedef __attribute__((ext_vector_type(16))) _Float16 v16h;
typedef __attribute__((ext_vector_type(8)))  float    v8f;

#define B_      256
#define ROUTES_ 512
#define NCAPS_  10

__device__ __forceinline__ float act_apply(float v, int act) {
  if (act == 1) return fmaxf(v, 0.0f);
  if (act == 2) return 1.0f / (1.0f + expf(-v));
  return v;
}

// ---------------------------------------------------------------------------
// Wave-level WMMA layouts per CDNA5 ISA 7.12.2 (wave32):
//  A (16x32 f16): lanes 0-15 -> M=lane, halves 0..7 = K 0..7, 8..15 = K 16..23
//                 lanes 16-31 -> M=lane-16, halves = K 8..15 / K 24..31
//  B (32x16 f16), this code's contiguous k (A loaded alike): lanes 0-15 -> N=lane, halves 0..15 = K 0..15
//                 lanes 16-31 -> N=lane-16, halves = K 16..31
//  C/D (16x16 f32): VGPR g: lanes 0-15 -> M=g, lanes 16-31 -> M=g+8; N=lane%16
// ---------------------------------------------------------------------------

// Generic dense GEMM: C[m,n] = act(sum_k A[m,k] * W[n,k] + bias[n])
// Block tile 32(M) x 128(N), 8 waves (2x4), K-step 32. M%32==0, N%128==0, K%32==0.
// blockIdx.x = m-tile (fast-varying -> weight-tile reuse in L2), blockIdx.y = n-tile.
__global__ __launch_bounds__(256) void k_gemm_wmma(
    const float* __restrict__ A, const float* __restrict__ W,
    const float* __restrict__ bias, float* __restrict__ C,
    int M, int Ncols, int K, int act)
{
  __shared__ _Float16 As[32][34];   // 34-half row stride: 17 banks, conflict-free
  __shared__ _Float16 Bs[128][34];
  const int tid  = threadIdx.x;
  const int lane = tid & 31;
  const int wave = tid >> 5;
  const int wm = wave >> 2, wn = wave & 3;
  const int m0 = blockIdx.x * 32;
  const int n0 = blockIdx.y * 128;
  v8f acc0 = {}; v8f acc1 = {};
  for (int k0 = 0; k0 < K; k0 += 32) {
    for (int e = tid; e < 32 * 32; e += 256) {
      int r = e >> 5, kk = e & 31;
      As[r][kk] = (_Float16)A[(size_t)(m0 + r) * K + k0 + kk];
    }
    for (int e = tid; e < 128 * 32; e += 256) {
      int r = e >> 5, kk = e & 31;
      Bs[r][kk] = (_Float16)W[(size_t)(n0 + r) * K + k0 + kk];
    }
    __syncthreads();
    const int arow = (lane & 15) + wm * 16;
    const int ak = (lane < 16) ? 0 : 16;
    v16h af;
#pragma unroll
    for (int j = 0; j < 16; ++j) { af[j] = As[arow][ak + j]; }
    const int bk = (lane < 16) ? 0 : 16;
    const int bc = wn * 32 + (lane & 15);
    v16h bf0, bf1;
#pragma unroll
    for (int j = 0; j < 16; ++j) { bf0[j] = Bs[bc][bk + j]; bf1[j] = Bs[bc + 16][bk + j]; }
    acc0 = __builtin_amdgcn_wmma_f32_16x16x32_f16(false, af, false, bf0, (short)0, acc0, false, false);
    acc1 = __builtin_amdgcn_wmma_f32_16x16x32_f16(false, af, false, bf1, (short)0, acc1, false, false);
    __syncthreads();
  }
  const int mbase = m0 + wm * 16 + ((lane < 16) ? 0 : 8);
  const int nb0 = n0 + wn * 32 + (lane & 15);
#pragma unroll
  for (int g = 0; g < 8; ++g) {
    int m = mbase + g;
    C[(size_t)m * Ncols + nb0]      = act_apply(acc0[g] + bias[nb0], act);
    C[(size_t)m * Ncols + nb0 + 16] = act_apply(acc1[g] + bias[nb0 + 16], act);
  }
}

// conv1 as implicit-im2col GEMM: M = B*144 (row m = b*144 + py*12 + px),
// N = 256 out-channels, K = 81 padded to 96. ReLU epilogue.
// Stores h[(b*256+oc)*144 + p] (channel-major for the primary-caps loader).
__global__ __launch_bounds__(256) void k_conv1_wmma(
    const float* __restrict__ data, const float* __restrict__ w,
    const float* __restrict__ bias, float* __restrict__ h)
{
  __shared__ _Float16 As[32][34];
  __shared__ _Float16 Bs[128][34];
  const int tid  = threadIdx.x;
  const int lane = tid & 31;
  const int wave = tid >> 5;
  const int wm = wave >> 2, wn = wave & 3;
  const int m0 = blockIdx.y * 32;
  const int n0 = blockIdx.x * 128;
  v8f acc0 = {}; v8f acc1 = {};
  for (int k0 = 0; k0 < 96; k0 += 32) {
    for (int e = tid; e < 32 * 32; e += 256) {
      int r = e >> 5, kk = e & 31;
      int m = m0 + r;
      int b = m / 144, p = m % 144;
      int py = p / 12, px = p % 12;
      int k = k0 + kk;
      float val = 0.0f;
      if (k < 81) {
        int dy = k / 9, dx = k % 9;
        val = data[(size_t)b * 400 + (py + dy) * 20 + (px + dx)];
      }
      As[r][kk] = (_Float16)val;
    }
    for (int e = tid; e < 128 * 32; e += 256) {
      int r = e >> 5, kk = e & 31;
      int k = k0 + kk;
      Bs[r][kk] = (_Float16)((k < 81) ? w[(size_t)(n0 + r) * 81 + k] : 0.0f);
    }
    __syncthreads();
    const int arow = (lane & 15) + wm * 16;
    const int ak = (lane < 16) ? 0 : 16;
    v16h af;
#pragma unroll
    for (int j = 0; j < 16; ++j) { af[j] = As[arow][ak + j]; }
    const int bk = (lane < 16) ? 0 : 16;
    const int bc = wn * 32 + (lane & 15);
    v16h bf0, bf1;
#pragma unroll
    for (int j = 0; j < 16; ++j) { bf0[j] = Bs[bc][bk + j]; bf1[j] = Bs[bc + 16][bk + j]; }
    acc0 = __builtin_amdgcn_wmma_f32_16x16x32_f16(false, af, false, bf0, (short)0, acc0, false, false);
    acc1 = __builtin_amdgcn_wmma_f32_16x16x32_f16(false, af, false, bf1, (short)0, acc1, false, false);
    __syncthreads();
  }
  const int mbase = m0 + wm * 16 + ((lane < 16) ? 0 : 8);
  const int nb0 = n0 + wn * 32 + (lane & 15);
#pragma unroll
  for (int g = 0; g < 8; ++g) {
    int m = mbase + g;
    int b = m / 144, p = m % 144;
    h[((size_t)b * 256 + nb0) * 144 + p]      = fmaxf(acc0[g] + bias[nb0], 0.0f);
    h[((size_t)b * 256 + nb0 + 16) * 144 + p] = fmaxf(acc1[g] + bias[nb0 + 16], 0.0f);
  }
}

// PrimaryCaps conv as implicit-im2col GEMM:
// M = B*16 (row m = b*16 + py*4 + px), N = 256 oc, K = 256*81 = 20736 (=648*32).
// 21 MB of weights stay L2-resident (192 MB L2) and are reused across batches.
// Stores u[(b*256+oc)*16 + p] (no activation).
__global__ __launch_bounds__(256) void k_prim_wmma(
    const float* __restrict__ h, const float* __restrict__ w,
    const float* __restrict__ bias, float* __restrict__ u)
{
  __shared__ _Float16 As[32][34];
  __shared__ _Float16 Bs[128][34];
  const int tid  = threadIdx.x;
  const int lane = tid & 31;
  const int wave = tid >> 5;
  const int wm = wave >> 2, wn = wave & 3;
  const int m0 = blockIdx.y * 32;
  const int n0 = blockIdx.x * 128;
  v8f acc0 = {}; v8f acc1 = {};
  for (int k0 = 0; k0 < 20736; k0 += 32) {
    for (int e = tid; e < 32 * 32; e += 256) {
      int r = e >> 5, kk = e & 31;
      int m = m0 + r;
      int b = m >> 4, p = m & 15;
      int py = p >> 2, px = p & 3;
      int k = k0 + kk;
      int ic = k / 81, j = k % 81;
      int dy = j / 9, dx = j % 9;
      As[r][kk] = (_Float16)h[((size_t)b * 256 + ic) * 144 + (py + dy) * 12 + (px + dx)];
    }
    for (int e = tid; e < 128 * 32; e += 256) {
      int r = e >> 5, kk = e & 31;
      Bs[r][kk] = (_Float16)w[(size_t)(n0 + r) * 20736 + k0 + kk];
    }
    __syncthreads();
    const int arow = (lane & 15) + wm * 16;
    const int ak = (lane < 16) ? 0 : 16;
    v16h af;
#pragma unroll
    for (int j = 0; j < 16; ++j) { af[j] = As[arow][ak + j]; }
    const int bk = (lane < 16) ? 0 : 16;
    const int bc = wn * 32 + (lane & 15);
    v16h bf0, bf1;
#pragma unroll
    for (int j = 0; j < 16; ++j) { bf0[j] = Bs[bc][bk + j]; bf1[j] = Bs[bc + 16][bk + j]; }
    acc0 = __builtin_amdgcn_wmma_f32_16x16x32_f16(false, af, false, bf0, (short)0, acc0, false, false);
    acc1 = __builtin_amdgcn_wmma_f32_16x16x32_f16(false, af, false, bf1, (short)0, acc1, false, false);
    __syncthreads();
  }
  const int mbase = m0 + wm * 16 + ((lane < 16) ? 0 : 8);
  const int nb0 = n0 + wn * 32 + (lane & 15);
#pragma unroll
  for (int g = 0; g < 8; ++g) {
    int m = mbase + g;
    int b = m >> 4, p = m & 15;
    u[((size_t)b * 256 + nb0) * 16 + p]      = acc0[g] + bias[nb0];
    u[((size_t)b * 256 + nb0 + 16) * 16 + p] = acc1[g] + bias[nb0 + 16];
  }
}

// ---------------------------------------------------------------------------
// Scalar / reduction kernels
// ---------------------------------------------------------------------------

__global__ void k_zero(float* __restrict__ p, int n) {
  int i = blockIdx.x * 256 + threadIdx.x;
  if (i < n) p[i] = 0.0f;
}

// torch .view quirk: route r, dim i2 maps to flat f = r*8+i2 of (8,32,4,4):
// i = f/512, g = (f%512)/16, p = f%16 -> channel oc = i*32+g, position p.
// For fixed r: oc fixed, p = (r&1)*8 + i2 (8 consecutive positions).
__global__ void k_squash(const float* __restrict__ u, float* __restrict__ x) {
  int idx = blockIdx.x * 256 + threadIdx.x;       // b*512 + r
  if (idx >= B_ * ROUTES_) return;
  int b = idx >> 9, r = idx & 511;
  int f = r * 8;
  int i = f >> 9, rem = f & 511;
  int g = rem >> 4, p = rem & 15;
  int oc = i * 32 + g;
  const float* up = u + ((size_t)(b * 256 + oc) * 16 + p);
  float vv[8]; float sn = 0.0f;
#pragma unroll
  for (int j = 0; j < 8; ++j) { vv[j] = up[j]; sn += vv[j] * vv[j]; }
  float sc = sn / ((1.0f + sn) * sqrtf(sn));
  float* xp = x + (size_t)idx * 8;
#pragma unroll
  for (int j = 0; j < 8; ++j) xp[j] = vv[j] * sc;
}

// u_hat[b,r,c,o] = sum_i W[r,c,o,i] * x[b,r,i]   (memory-bound: 84MB write)
__global__ void k_uhat(const float* __restrict__ x, const float* __restrict__ W,
                       float* __restrict__ uhat) {
  size_t idx = (size_t)blockIdx.x * 256 + threadIdx.x; // ((b*512+r)*160+co)
  if (idx >= (size_t)B_ * ROUTES_ * 160) return;
  int co = (int)(idx % 160);
  size_t br = idx / 160;
  int r = (int)(br & 511);
  const float* xp = x + br * 8;
  const float* wp = W + ((size_t)r * 160 + co) * 8;
  float s = 0.0f;
#pragma unroll
  for (int i = 0; i < 8; ++i) s += wp[i] * xp[i];
  uhat[idx] = s;
}

// c[r,cls] = softmax over routes (axis 0) of b_ij; one block per class.
__global__ __launch_bounds__(256) void k_softmax_routes(
    const float* __restrict__ bij, float* __restrict__ cmat) {
  int cls = blockIdx.x, tid = threadIdx.x;
  __shared__ float red[256];
  float m = -INFINITY;
  for (int r = tid; r < ROUTES_; r += 256) m = fmaxf(m, bij[r * NCAPS_ + cls]);
  red[tid] = m; __syncthreads();
  for (int s = 128; s > 0; s >>= 1) { if (tid < s) red[tid] = fmaxf(red[tid], red[tid + s]); __syncthreads(); }
  m = red[0]; __syncthreads();
  float sum = 0.0f;
  for (int r = tid; r < ROUTES_; r += 256) sum += expf(bij[r * NCAPS_ + cls] - m);
  red[tid] = sum; __syncthreads();
  for (int s = 128; s > 0; s >>= 1) { if (tid < s) red[tid] += red[tid + s]; __syncthreads(); }
  float inv = 1.0f / red[0]; __syncthreads();
  for (int r = tid; r < ROUTES_; r += 256) cmat[r * NCAPS_ + cls] = expf(bij[r * NCAPS_ + cls] - m) * inv;
}

// s[b,c,:] = sum_r c[r,c]*u_hat[b,r,c,:]; v = elementwise squash(s).
// One block per (b,c); tid = rg*16 + o.
__global__ __launch_bounds__(256) void k_route_sv(
    const float* __restrict__ cmat, const float* __restrict__ uhat,
    float* __restrict__ v) {
  int bc = blockIdx.x;
  int b = bc / NCAPS_, cls = bc % NCAPS_;
  int tid = threadIdx.x;
  int o = tid & 15, rg = tid >> 4;
  __shared__ float red[256];
  float sum = 0.0f;
  for (int r = rg; r < ROUTES_; r += 16)
    sum += cmat[r * NCAPS_ + cls] * uhat[(((size_t)b * ROUTES_ + r) * NCAPS_ + cls) * 16 + o];
  red[tid] = sum; __syncthreads();
  for (int s = 128; s >= 16; s >>= 1) { if (tid < s) red[tid] += red[tid + s]; __syncthreads(); }
  if (tid < 16) {
    float s = red[tid];
    float sq = s * s;
    v[(size_t)bc * 16 + tid] = sq * s / ((1.0f + sq) * sqrtf(sq));
  }
}

// b_ij[r,c] += mean_b sum_o u_hat[b,r,c,o]*v[b,c,o]; one block per route.
__global__ __launch_bounds__(256) void k_route_update(
    const float* __restrict__ uhat, const float* __restrict__ v,
    float* __restrict__ bij) {
  int r = blockIdx.x, b = threadIdx.x;
  __shared__ float red[256];
  const float* uh = uhat + ((size_t)b * ROUTES_ + r) * 160;
  for (int cls = 0; cls < NCAPS_; ++cls) {
    float a = 0.0f;
#pragma unroll
    for (int o = 0; o < 16; ++o) a += uh[cls * 16 + o] * v[(size_t)(b * NCAPS_ + cls) * 16 + o];
    red[b] = a; __syncthreads();
    for (int s = 128; s > 0; s >>= 1) { if (b < s) red[b] += red[b + s]; __syncthreads(); }
    if (b == 0) bij[r * NCAPS_ + cls] += red[0] * (1.0f / 256.0f);
    __syncthreads();
  }
}

__global__ void k_norms(const float* __restrict__ v, float* __restrict__ cl) {
  int idx = blockIdx.x * 256 + threadIdx.x;   // b*10 + c
  if (idx >= B_ * NCAPS_) return;
  float s = 0.0f;
  const float* vp = v + (size_t)idx * 16;
#pragma unroll
  for (int o = 0; o < 16; ++o) s += vp[o] * vp[o];
  cl[idx] = sqrtf(s);
}

// legacy softmax over axis 0 (batch) per class -> logsumexp per class.
__global__ __launch_bounds__(256) void k_lse(const float* __restrict__ cl,
                                             float* __restrict__ lse) {
  int cls = blockIdx.x, b = threadIdx.x;
  __shared__ float red[256];
  float val = cl[b * NCAPS_ + cls];
  red[b] = val; __syncthreads();
  for (int s = 128; s > 0; s >>= 1) { if (b < s) red[b] = fmaxf(red[b], red[b + s]); __syncthreads(); }
  float m = red[0]; __syncthreads();
  red[b] = expf(val - m); __syncthreads();
  for (int s = 128; s > 0; s >>= 1) { if (b < s) red[b] += red[b + s]; __syncthreads(); }
  if (b == 0) lse[cls] = m + logf(red[0]);
}

// argmax over classes of (cl - lse) (monotonic with legacy softmax), mask v.
__global__ void k_mask(const float* __restrict__ cl, const float* __restrict__ lse,
                       const float* __restrict__ v, float* __restrict__ flat) {
  int b = blockIdx.x, t = threadIdx.x;   // t = c*16+o, 160 threads
  float best = -INFINITY; int bi = 0;
  for (int cc = 0; cc < NCAPS_; ++cc) {
    float sc = cl[b * NCAPS_ + cc] - lse[cc];
    if (sc > best) { best = sc; bi = cc; }
  }
  int cc = t >> 4;
  flat[(size_t)b * 160 + t] = (cc == bi) ? v[(size_t)b * 160 + t] : 0.0f;
}

// ---------------------------------------------------------------------------
extern "C" void kernel_launch(void* const* d_in, const int* in_sizes, int n_in,
                              void* d_out, int out_size, void* d_ws, size_t ws_size,
                              hipStream_t stream) {
  (void)in_sizes; (void)n_in; (void)out_size; (void)ws_size;
  const float* data    = (const float*)d_in[0];
  const float* conv1_w = (const float*)d_in[1];
  const float* conv1_b = (const float*)d_in[2];
  const float* prim_w  = (const float*)d_in[3];
  const float* prim_b  = (const float*)d_in[4];
  const float* W_digit = (const float*)d_in[5];
  const float* dec_w1  = (const float*)d_in[6];
  const float* dec_b1  = (const float*)d_in[7];
  const float* dec_w2  = (const float*)d_in[8];
  const float* dec_b2  = (const float*)d_in[9];
  const float* dec_w3  = (const float*)d_in[10];
  const float* dec_b3  = (const float*)d_in[11];
  float* out = (float*)d_out;

  // Workspace layout (bytes). u_hat aliases h+u, which are dead by then.
  char* ws = (char*)d_ws;
  const size_t OFF_UHAT = 0;                         // 83,886,080 B
  const size_t OFF_H    = 0;                         // 37,748,736 B (dead before u_hat)
  const size_t OFF_U    = 37748736;                  //  4,194,304 B (dead before u_hat)
  const size_t OFF_X    = 83886080;                  //  4,194,304 B
  const size_t OFF_V    = OFF_X   + 4194304;         //    163,840 B
  const size_t OFF_BIJ  = OFF_V   + 163840;          //     20,480 B
  const size_t OFF_C    = OFF_BIJ + 20480;           //     20,480 B
  const size_t OFF_CL   = OFF_C   + 20480;           //     10,240 B
  const size_t OFF_LSE  = OFF_CL  + 10240;           //        256 B
  const size_t OFF_FLAT = OFF_LSE + 256;             //    163,840 B
  const size_t OFF_H1   = OFF_FLAT + 163840;         //    524,288 B
  const size_t OFF_H2   = OFF_H1   + 524288;         //  1,048,576 B  (end ~90MB)

  float* h    = (float*)(ws + OFF_H);
  float* u    = (float*)(ws + OFF_U);
  float* uhat = (float*)(ws + OFF_UHAT);
  float* x    = (float*)(ws + OFF_X);
  float* v    = (float*)(ws + OFF_V);
  float* bij  = (float*)(ws + OFF_BIJ);
  float* cmat = (float*)(ws + OFF_C);
  float* cl   = (float*)(ws + OFF_CL);
  float* lse  = (float*)(ws + OFF_LSE);
  float* flat = (float*)(ws + OFF_FLAT);
  float* h1   = (float*)(ws + OFF_H1);
  float* h2   = (float*)(ws + OFF_H2);

  // 1) conv1: M=36864, N=256, K=81->96
  k_conv1_wmma<<<dim3(2, 1152), 256, 0, stream>>>(data, conv1_w, conv1_b, h);
  // 2) primary caps: M=4096, N=256, K=20736
  k_prim_wmma<<<dim3(2, 128), 256, 0, stream>>>(h, prim_w, prim_b, u);
  // 3) squash -> x [B,512,8]
  k_squash<<<512, 256, 0, stream>>>(u, x);
  // 4) u_hat [B,512,10,16]
  k_uhat<<<81920, 256, 0, stream>>>(x, W_digit, uhat);
  // 5) routing (3 iterations, b_ij shared across batch)
  k_zero<<<20, 256, 0, stream>>>(bij, ROUTES_ * NCAPS_);
  for (int it = 0; it < 3; ++it) {
    k_softmax_routes<<<NCAPS_, 256, 0, stream>>>(bij, cmat);
    k_route_sv<<<B_ * NCAPS_, 256, 0, stream>>>(cmat, uhat, v);
    if (it < 2) k_route_update<<<ROUTES_, 256, 0, stream>>>(uhat, v, bij);
  }
  // 6) classes, legacy axis-0 softmax, argmax, mask
  k_norms<<<10, 256, 0, stream>>>(v, cl);
  k_lse<<<NCAPS_, 256, 0, stream>>>(cl, lse);
  k_mask<<<B_, 160, 0, stream>>>(cl, lse, v, flat);
  // 7) decoder (grid: x = m-tiles of 8, y = n-tiles -> weight reuse in L2)
  k_gemm_wmma<<<dim3(8, 4),   256, 0, stream>>>(flat, dec_w1, dec_b1, h1, 256, 512, 160, 1);
  k_gemm_wmma<<<dim3(8, 8),   256, 0, stream>>>(h1,   dec_w2, dec_b2, h2, 256, 1024, 512, 1);
  k_gemm_wmma<<<dim3(8, 800), 256, 0, stream>>>(h2,   dec_w3, dec_b3, out, 256, 102400, 1024, 2);
}
